// DepthRelationEmbedding_74887049773205
// MI455X (gfx1250) — matrix operand order repaired and checked by reading, not compiled, not run on hardware
//
#include <hip/hip_runtime.h>

// DepthRelationEmbedding, restructured via sin/cos angle-difference identities
// into a single 2400x256x1920 f16->f32 WMMA GEMM.
//
//   out[h*300+n, p] = relu( b[h] + sum_k A[m,k] * B[k,p] )
//   A[m,k<128]   = sin(a_n s_k) W[2k,h] + cos(a_n s_k) W[2k+1,h]
//   A[m,128+k]   = sin(a_n s_k) W[2k+1,h] - cos(a_n s_k) W[2k,h]
//   B[k<128, p]  = cos(b_p s_k),   B[128+k, p] = sin(b_p s_k)
//   a_n = log(relu(pd[n])+eps), b_p = log(dm[p]+eps), s_k = 100*10000^(-k/128)
//
// Transcendentals drop 147M -> 0.57M; the GEMM (2.36 GFLOP) runs on
// v_wmma_f32_16x16x32_f16 with both factor matrices L2-resident (2.2 MB of
// 192 MB), so the kernel is bound by the 18.4 MB f32 output (~0.8us @ 23.3TB/s).

typedef __attribute__((ext_vector_type(16))) _Float16 v16h;
typedef __attribute__((ext_vector_type(8)))  float    v8f;

#define NPTS   300
#define HEADS  8
#define HWDIM  1920
#define MDIM   (NPTS * HEADS)   // 2400
#define KDIM   256
#define MT     (MDIM / 16)      // 150 row tiles
#define NT     (HWDIM / 16)     // 120 col tiles
#define KT     (KDIM / 32)      // 8 K steps of 32
#define NTPW   4                // N-tiles per wave (A-fragment reuse)
#define NGRP   (NT / NTPW)      // 30 strip groups
#define NWAVES (MT * NGRP)      // 4500 waves
#define EPS_F  1e-5f
#define SCALE_F 100.0f
// ln(10000)/128
#define LN1E4_OVER_128 0.071920518112945211f

__device__ __forceinline__ float freq_scale(int k) {
    // s_k = 100 * 10000^(-k/128)
    return SCALE_F * expf(-LN1E4_OVER_128 * (float)k);
}

// ---- Stage A: build A (2400 x 256) in WMMA A-fragment order (f16) ----------
// A 16x32 f16 fragment layout (ISA 7.12.2): for row r (0..15), kk (0..31):
//   g = kk/8 ; lane = r + 16*(g&1) ; half-slot = (kk&7) + 8*(g>>1)
// One tile = 32 lanes * 16 halves = 512 halves -> GEMM lane load is one
// contiguous 32B v16h per K-step.
__global__ __launch_bounds__(256) void dre_stage_a(
    const float* __restrict__ pd, const float* __restrict__ W,
    _Float16* __restrict__ Ast)
{
    int idx = blockIdx.x * 256 + threadIdx.x;   // MDIM * 256 threads
    int m  = idx >> 8;                          // 0..2399
    int k2 = idx & 255;                         // 0..255
    int n = m % NPTS;
    int h = m / NPTS;
    int k = k2 & 127;

    float s = freq_scale(k);
    float a = logf(fmaxf(pd[n], 0.0f) + EPS_F);   // relu + eps + log
    float sa, ca;
    sincosf(a * s, &sa, &ca);

    float Ws = W[(2 * k)     * HEADS + h];
    float Wc = W[(2 * k + 1) * HEADS + h];
    float val = (k2 < 128) ? (sa * Ws + ca * Wc) : (sa * Wc - ca * Ws);

    int mt = m >> 4, r  = m & 15;
    int kt = k2 >> 5, kk = k2 & 31, g = kk >> 3;
    int lane = r + ((g & 1) << 4);
    int pos  = (kk & 7) + ((g >> 1) << 3);
    Ast[(size_t)(mt * KT + kt) * 512 + lane * 16 + pos] = (_Float16)val;
}

// ---- Stage B: build B (256 x 1920) in WMMA B-fragment order (f16) ----------
// B 32x16 f16 fragment layout (same k order as A): for column c, kk (0..31):
//   g = kk/8 ; lane = c + 16*(g&1) ; half-slot = (kk&7) + 8*(g>>1)
__global__ __launch_bounds__(256) void dre_stage_b(
    const float* __restrict__ dm, _Float16* __restrict__ Bst)
{
    int idx = blockIdx.x * 256 + threadIdx.x;   // KDIM * HWDIM threads
    int k2 = idx / HWDIM;
    int p  = idx - k2 * HWDIM;
    int k  = k2 & 127;

    float s  = freq_scale(k);
    float bp = logf(dm[p] + EPS_F);
    float sb, cb;
    sincosf(bp * s, &sb, &cb);
    float val = (k2 < 128) ? cb : sb;

    int nt = p >> 4, c  = p & 15;
    int kt = k2 >> 5, kk = k2 & 31;
    int lane = c + (((kk >> 3) & 1) << 4);
    int pos  = (kk & 7) + ((kk >> 4) << 3);
    Bst[(size_t)(kt * NT + nt) * 512 + lane * 16 + pos] = (_Float16)val;
}

// ---- GEMM: one wave per 16x64 output strip (1 M-tile x 4 N-tiles) ----------
// Per K-step: 1 A-fragment load + 4 B-fragment loads all issued before the
// 4 WMMAs consume them (stepped s_wait_loadcnt instead of load->wait->wmma).
__global__ __launch_bounds__(256) void dre_wmma_gemm(
    const _Float16* __restrict__ Ast, const _Float16* __restrict__ Bst,
    const float* __restrict__ bias, float* __restrict__ out)
{
    int wid = blockIdx.x * 8 + (threadIdx.x >> 5);
    if (wid >= NWAVES) return;                  // wave-uniform guard: EXEC
    int lane = threadIdx.x & 31;                // stays all-1s for WMMA waves
    int mt  = wid / NGRP;
    int nt0 = (wid - mt * NGRP) * NTPW;

    const _Float16* aptr = Ast + (size_t)(mt * KT) * 512 + lane * 16;
    const _Float16* bptr = Bst + (size_t)nt0 * 512 + lane * 16;

    v8f acc[NTPW] = {};
#pragma unroll
    for (int kt = 0; kt < KT; ++kt) {
        // Issue all 5 fragment loads for this K-step up front.
        v16h af = *(const v16h*)(aptr + (size_t)kt * 512);
        v16h bf[NTPW];
#pragma unroll
        for (int j = 0; j < NTPW; ++j)
            bf[j] = *(const v16h*)(bptr + (size_t)(kt * NT + j) * 512);
        // Then drain them through the matrix pipe.
#pragma unroll
        for (int j = 0; j < NTPW; ++j)
            acc[j] = __builtin_amdgcn_wmma_f32_16x16x32_f16(
                /*neg_a=*/false, af, /*neg_b=*/false, bf[j],
                /*c_mod=*/(short)0, acc[j], /*reuse_a=*/false, /*reuse_b=*/false);
    }

    // D fragment (f32 16x16): VGPR r <-> M = r + 8*(lane/16), N = lane%16.
    // h = m/300 changes at most once inside a 16-row tile: one division,
    // then a compare-add per row (replaces 8 magic-number division chains).
    int h0  = (mt * 16) / NPTS;
    int hb  = (h0 + 1) * NPTS;                  // head boundary (>= 2400 is safe)
    int nc0   = nt0 * 16 + (lane & 15);
    int mbase = mt * 16 + ((lane >> 4) << 3);
#pragma unroll
    for (int r = 0; r < 8; ++r) {
        int m = mbase + r;
        int h = h0 + (m >= hb ? 1 : 0);
        float bv = bias[h];
#pragma unroll
        for (int j = 0; j < NTPW; ++j) {
            out[(size_t)m * HWDIM + nc0 + j * 16] = fmaxf(acc[j][r] + bv, 0.0f);
        }
    }
}

extern "C" void kernel_launch(void* const* d_in, const int* in_sizes, int n_in,
                              void* d_out, int out_size, void* d_ws, size_t ws_size,
                              hipStream_t stream)
{
    const float* pd   = (const float*)d_in[0];  // (1, 300)
    const float* dm   = (const float*)d_in[1];  // (1, 24, 80) -> flat 1920
    const float* W    = (const float*)d_in[2];  // (256, 8) row-major
    const float* bias = (const float*)d_in[3];  // (8,)
    float* out = (float*)d_out;                 // (8*300, 1920) f32

    // Workspace: A staged 2400*256 halves = 1,228,800 B; B staged 256*1920
    // halves = 983,040 B. Both L2-resident during the GEMM.
    _Float16* Ast = (_Float16*)d_ws;
    _Float16* Bst = (_Float16*)((char*)d_ws + (size_t)MDIM * KDIM * 2);

    dre_stage_a<<<MDIM, 256, 0, stream>>>(pd, W, Ast);
    dre_stage_b<<<(KDIM * HWDIM) / 256, 256, 0, stream>>>(dm, Bst);
    dre_wmma_gemm<<<(NWAVES + 7) / 8, 256, 0, stream>>>(Ast, Bst, bias, out);
}
